// MemFullAttention_19585050870313
// MI455X (gfx1250) — compile-verified, matrix operand order checked by reading, not run on hardware
//
#include <hip/hip_runtime.h>
#include <hip/hip_bf16.h>

// ---------------------------------------------------------------------------
// MemFullAttention for MI455X (gfx1250), bf16 WMMA pipeline.
//   1) cvt: Wqkv, Wout f32 -> bf16 (workspace)
//   2) gemm<0>: Q   = x · Wqkv[:, 0:1024]      (only the 11264 needed q-tokens)
//   3) gemm<1>: K,V = x · Wqkv[:, 1024:3072]   (only the 24576 needed kv-tokens)
//   4) attn: 44 problems x 16 heads, 256 q x 768 kv, flash-style online softmax
//   5) gemm<2>: out = O · Wout + bout          (f32 output)
// All matmuls use v_wmma_f32_16x16x32_bf16; LDS staging uses the gfx1250
// async global->LDS path (ASYNCcnt) when the toolchain exposes it.
// ---------------------------------------------------------------------------

typedef __bf16 bf16_t;
typedef __attribute__((ext_vector_type(16))) __bf16        v16bf;
typedef __attribute__((ext_vector_type(8)))  __bf16        v8bf;
typedef __attribute__((ext_vector_type(8)))  float         v8f;
typedef __attribute__((ext_vector_type(4)))  float         v4f;
typedef __attribute__((ext_vector_type(2)))  unsigned int  v2u;
typedef __attribute__((ext_vector_type(4)))  int           v4i;
typedef __attribute__((ext_vector_type(2)))  int           v2i;

#if __has_builtin(__builtin_amdgcn_global_load_async_to_lds_b128) && \
    __has_builtin(__builtin_amdgcn_global_load_async_to_lds_b64)  && \
    __has_builtin(__builtin_amdgcn_s_wait_asynccnt)
#define USE_ASYNC_LDS 1
#else
#define USE_ASYNC_LDS 0
#endif

#if USE_ASYNC_LDS
typedef __attribute__((address_space(1))) v4i g_v4i;
typedef __attribute__((address_space(3))) v4i l_v4i;
typedef __attribute__((address_space(1))) v2i g_v2i;
typedef __attribute__((address_space(3))) v2i l_v2i;
// DMA 16 bytes global -> LDS, tracked by ASYNCcnt.
__device__ __forceinline__ void async_b128(void* lds_dst, const void* gsrc) {
  __builtin_amdgcn_global_load_async_to_lds_b128(
      (g_v4i*)(v4i*)gsrc, (l_v4i*)(v4i*)lds_dst, 0, 0);
}
__device__ __forceinline__ void async_b64(void* lds_dst, const void* gsrc) {
  __builtin_amdgcn_global_load_async_to_lds_b64(
      (g_v2i*)(v2i*)gsrc, (l_v2i*)(v2i*)lds_dst, 0, 0);
}
__device__ __forceinline__ void async_wait0() {
  __builtin_amdgcn_s_wait_asynccnt(0);
}
#endif

// Problem constants (from reference): B=4, T=32, S=H*W=256, D=1024,
// heads=16, head_dim=64, length=4, n_chunks=8, slices=11.
constexpr int NQROW  = 4 * 11 * 256;    // 11264 query tokens
constexpr int NKVROW = 4 * 8 * 3 * 256; // 24576 kv tokens

__device__ __forceinline__ int qrow_to_xrow(int r) {
  // r = ((b*11)+slice)*256 + sp  ->  x row ((b*32)+t)*256 + sp
  int sp = r & 255; int u = r >> 8;
  int b = u / 11, slice = u % 11;
  int t = (slice < 7) ? (slice * 4) : (28 + (slice - 7));
  return ((b * 32 + t) << 8) + sp;
}
__device__ __forceinline__ int kvrow_to_xrow(int j) {
  // j = ((b*24) + chunk*3 + (slot-1))*256 + sp
  int sp = j & 255; int u = j >> 8;
  int b = u / 24, v = u % 24;
  int chunk = v / 3, slot = (v % 3) + 1;
  return ((b * 32 + chunk * 4 + slot) << 8) + sp;
}

// ---------------------------------------------------------------------------
// 16x32 bf16 operand fragment loader, ISA A-matrix layout (CDNA5 ISA):
//   lanes 0-15:  row=lane,    elems 0..7 = K0..7,  elems 8..15 = K16..23
//   lanes 16-31: row=lane-16, elems 0..7 = K8..15, elems 8..15 = K24..31
// Row-major [row][k] source => each half is one contiguous b128 load.
// Also serves the B operand when B is staged transposed [n][k].
// ---------------------------------------------------------------------------
__device__ __forceinline__ v16bf load_frag(const bf16_t* base, int stride,
                                           int row0, int k0, int lane) {
  int r = row0 + (lane & 15);
  const bf16_t* p = base + (size_t)r * stride + k0 + ((lane >> 4) << 3);
  v8bf lo = *(const v8bf*)(p);
  v8bf hi = *(const v8bf*)(p + 16);
  return __builtin_shufflevector(lo, hi, 0, 1, 2, 3, 4, 5, 6, 7,
                                 8, 9, 10, 11, 12, 13, 14, 15);
}

__device__ __forceinline__ v8f wmma_bf16(v16bf a, v16bf b, v8f c) {
  return __builtin_amdgcn_wmma_f32_16x16x32_bf16(
      /*neg_a=*/false, a, /*neg_b=*/false, b,
      /*c_mod=*/(short)0, c, /*reuse_a=*/false, /*reuse_b=*/false);
}

// ---------------------------------------------------------------------------
__global__ void cvt_f32_bf16(const float* __restrict__ in,
                             bf16_t* __restrict__ out, int n) {
  int i = blockIdx.x * blockDim.x + threadIdx.x;
  int stride = gridDim.x * blockDim.x;
  for (; i < n; i += stride) out[i] = (bf16_t)in[i];
}

// ---------------------------------------------------------------------------
// Tiled GEMM: C[M x N] = A[M x 1024] * Bw[1024 x N'] (bf16 -> f32 acc).
// Block tile 128x128, BK=32, 256 threads = 8 waves in a 4(M) x 2(N) grid;
// each wave owns 32x64 = 2x4 WMMA accumulators.
// MODE 0: A = x(f32) via qrow map, C -> Qbuf (bf16)
// MODE 1: A = x(f32) via kvrow map, C -> Kbuf / Vbuf split at n=1024 (bf16)
// MODE 2: A = Obuf(bf16) async-staged, C -> d_out (f32) + bias
// ---------------------------------------------------------------------------
template <int MODE>
__global__ __launch_bounds__(256) void gemm_bf16(
    const float* __restrict__ Af32, const bf16_t* __restrict__ Abf,
    const bf16_t* __restrict__ Bw, int ldb, int ncol0,
    bf16_t* __restrict__ C0, bf16_t* __restrict__ C1,
    float* __restrict__ Cf, const float* __restrict__ bias) {
  __shared__ bf16_t As[128][40];  // [m][k], pad to 40 (80B) -> conflict-free b128
  __shared__ bf16_t Bs[128][40];  // [n][k] (transposed stage)

  const int tid = threadIdx.x;
  const int lane = tid & 31, wave = tid >> 5;
  const int wm = wave >> 1, wn = wave & 1;  // 4 x 2 wave grid
  const int m0 = blockIdx.x * 128;
  const int n0 = blockIdx.y * 128;

  v8f acc[2][4];
#pragma unroll
  for (int mt = 0; mt < 2; ++mt)
#pragma unroll
    for (int nt = 0; nt < 4; ++nt) acc[mt][nt] = (v8f){};

  for (int k0 = 0; k0 < 1024; k0 += 32) {
    // ---- stage A tile ----
    {
      const int kseg = (tid & 7) << 2;  // 0,4,...,28
#pragma unroll
      for (int pass = 0; pass < 4; ++pass) {
        int rl = (tid >> 3) + pass * 32;
        int rg = m0 + rl;
        if (MODE == 2) {
          const bf16_t* src = Abf + (size_t)rg * 1024 + k0 + kseg;
#if USE_ASYNC_LDS
          async_b64(&As[rl][kseg], src);  // 4 bf16, DMA to LDS
#else
          *(v2u*)(&As[rl][kseg]) = *(const v2u*)src;
#endif
        } else {
          int xr = (MODE == 0) ? qrow_to_xrow(rg) : kvrow_to_xrow(rg);
          v4f f = *(const v4f*)(Af32 + (size_t)xr * 1024 + k0 + kseg);
          union { v2u u; bf16_t h[4]; } pk;
          pk.h[0] = (bf16_t)f[0]; pk.h[1] = (bf16_t)f[1];
          pk.h[2] = (bf16_t)f[2]; pk.h[3] = (bf16_t)f[3];
          *(v2u*)(&As[rl][kseg]) = pk.u;
        }
      }
    }
    // ---- stage B tile transposed: Bs[n][k] <- Bw[k0+kr][ncol0+n0+n] ----
    {
      const int kr = tid & 31;
      const int ns = tid >> 5;  // 0..7, 16 columns each
      const bf16_t* src = Bw + (size_t)(k0 + kr) * ldb + ncol0 + n0 + ns * 16;
#pragma unroll
      for (int i = 0; i < 16; ++i) Bs[ns * 16 + i][kr] = src[i];
    }
#if USE_ASYNC_LDS
    if (MODE == 2) async_wait0();  // A-tile DMA landed in LDS
#endif
    __syncthreads();

    // ---- 8 WMMAs per wave per k-step ----
    v16bf a[2], b[4];
#pragma unroll
    for (int mt = 0; mt < 2; ++mt)
      a[mt] = load_frag(&As[0][0], 40, wm * 32 + mt * 16, 0, lane);
#pragma unroll
    for (int nt = 0; nt < 4; ++nt)
      b[nt] = load_frag(&Bs[0][0], 40, wn * 64 + nt * 16, 0, lane);
#pragma unroll
    for (int mt = 0; mt < 2; ++mt)
#pragma unroll
      for (int nt = 0; nt < 4; ++nt)
        acc[mt][nt] = wmma_bf16(a[mt], b[nt], acc[mt][nt]);
    __syncthreads();
  }

  // ---- epilogue: C layout VGPR g -> row g + 8*(lane>=16), col lane%16 ----
  const int hi = lane >> 4, nn = lane & 15;
#pragma unroll
  for (int mt = 0; mt < 2; ++mt)
#pragma unroll
    for (int nt = 0; nt < 4; ++nt)
#pragma unroll
      for (int g = 0; g < 8; ++g) {
        int r = m0 + wm * 32 + mt * 16 + g + hi * 8;
        int c = n0 + wn * 64 + nt * 16 + nn;
        float v = acc[mt][nt][g];
        if (MODE == 0) {
          C0[(size_t)r * 1024 + c] = (bf16_t)v;
        } else if (MODE == 1) {
          if (c < 1024) C0[(size_t)r * 1024 + c] = (bf16_t)v;
          else          C1[(size_t)r * 1024 + (c - 1024)] = (bf16_t)v;
        } else {
          Cf[(size_t)r * 1024 + c] = v + bias[c];
        }
      }
}

// ---------------------------------------------------------------------------
// Flash attention: one block per (b, slice, head); 256 q rows x 768 kv,
// d = 64, scale = 1/8. 8 waves, each owns 32 q rows (2 M tiles).
// K streamed via async global->LDS DMA; V transposed in flight (sync);
// online softmax in registers.
// ---------------------------------------------------------------------------
__global__ __launch_bounds__(256) void attn_kernel(
    const bf16_t* __restrict__ Q, const bf16_t* __restrict__ K,
    const bf16_t* __restrict__ V, bf16_t* __restrict__ O) {
  __shared__ bf16_t Ks[128][72];       // [kv][d]  (B operand for Q·K^T)
  __shared__ bf16_t Vt[64][136];       // [d][kv]  (B operand for P·V)
  __shared__ bf16_t Ps[8][32][40];     // per-wave P restage (C->A layout)

  const int tid = threadIdx.x, lane = tid & 31, wave = tid >> 5;
  const int hi = lane >> 4, nn = lane & 15;
  const int head = blockIdx.x & 15;
  const int s2 = blockIdx.x >> 4;
  const int slice = s2 % 11, b = s2 / 11;
  const int chunk = (slice < 7) ? slice : 7;
  const int qbase = (b * 11 + slice) * 256;
  const int kvbase = (b * 24 + chunk * 3) * 256;
  const int dcol = head * 64;
  const float scale = 0.125f;  // 1/sqrt(64)

  // Q fragments straight from global (A layout matches row-major [token][d]).
  v16bf qf[2][2];
#pragma unroll
  for (int mt = 0; mt < 2; ++mt)
#pragma unroll
    for (int ks = 0; ks < 2; ++ks)
      qf[mt][ks] = load_frag(Q, 1024, qbase + wave * 32 + mt * 16,
                             dcol + ks * 32, lane);

  v8f acc[2][4];
  float mrow[2][8], lrow[2][8];
#pragma unroll
  for (int mt = 0; mt < 2; ++mt) {
#pragma unroll
    for (int dt = 0; dt < 4; ++dt) acc[mt][dt] = (v8f){};
#pragma unroll
    for (int g = 0; g < 8; ++g) { mrow[mt][g] = -1e30f; lrow[mt][g] = 0.f; }
  }

  for (int kb = 0; kb < 768; kb += 128) {
    __syncthreads();  // all waves done reading previous block
    // ---- stage K block [128][64]: contiguous 16B chunks -> async DMA ----
    {
      int row = tid >> 1;
      int seg = (tid & 1) * 32;
      const bf16_t* src = K + (size_t)(kvbase + kb + row) * 1024 + dcol + seg;
#if USE_ASYNC_LDS
#pragma unroll
      for (int i = 0; i < 4; ++i)
        async_b128(&Ks[row][seg + i * 8], src + i * 8);
#else
#pragma unroll
      for (int i = 0; i < 4; ++i)
        *(v8bf*)(&Ks[row][seg + i * 8]) = *(const v8bf*)(src + i * 8);
#endif
    }
    // ---- stage V block transposed: Vt[d][kv] (sync; transpose in flight) ----
    {
      int kv = tid & 127;
      int d0 = (tid >> 7) * 32;
      const bf16_t* src = V + (size_t)(kvbase + kb + kv) * 1024 + dcol + d0;
      v8bf vv[4];
#pragma unroll
      for (int i = 0; i < 4; ++i) vv[i] = *(const v8bf*)(src + i * 8);
#pragma unroll
      for (int i = 0; i < 32; ++i) Vt[d0 + i][kv] = vv[i >> 3][i & 7];
      // Prefetch next block's K/V rows toward L2 while we compute this one.
      if (kb + 128 < 768) {
        __builtin_prefetch(V + (size_t)(kvbase + kb + 128 + kv) * 1024 + dcol, 0, 1);
        __builtin_prefetch(K + (size_t)(kvbase + kb + 128 + kv) * 1024 + dcol, 0, 1);
      }
    }
#if USE_ASYNC_LDS
    async_wait0();  // K-block DMA landed in LDS
#endif
    __syncthreads();

    for (int sub = 0; sub < 4; ++sub) {  // 32-kv sub-blocks
      // ---- S = Q · K^T (2M x 2N tiles, K=64 -> 2 wmma each) ----
      v8f s[2][2];
#pragma unroll
      for (int mt = 0; mt < 2; ++mt)
#pragma unroll
        for (int nt = 0; nt < 2; ++nt) s[mt][nt] = (v8f){};
      v16bf kf[2][2];
#pragma unroll
      for (int nt = 0; nt < 2; ++nt)
#pragma unroll
        for (int ks = 0; ks < 2; ++ks)
          kf[nt][ks] = load_frag(&Ks[0][0], 72, sub * 32 + nt * 16,
                                 ks * 32, lane);
#pragma unroll
      for (int mt = 0; mt < 2; ++mt)
#pragma unroll
        for (int nt = 0; nt < 2; ++nt)
#pragma unroll
          for (int ks = 0; ks < 2; ++ks)
            s[mt][nt] = wmma_bf16(qf[mt][ks], kf[nt][ks], s[mt][nt]);

      // ---- online softmax update (row groups live in 16-lane halves) ----
#pragma unroll
      for (int mt = 0; mt < 2; ++mt) {
        float pm[8];
#pragma unroll
        for (int g = 0; g < 8; ++g)
          pm[g] = fmaxf(s[mt][0][g], s[mt][1][g]) * scale;
#pragma unroll
        for (int off = 1; off < 16; off <<= 1)
#pragma unroll
          for (int g = 0; g < 8; ++g)
            pm[g] = fmaxf(pm[g], __shfl_xor(pm[g], off, 32));
        float al[8];
#pragma unroll
        for (int g = 0; g < 8; ++g) {
          float mn = fmaxf(mrow[mt][g], pm[g]);
          al[g] = __expf(mrow[mt][g] - mn);
          mrow[mt][g] = mn;
        }
        float rs[8];
#pragma unroll
        for (int g = 0; g < 8; ++g) rs[g] = 0.f;
#pragma unroll
        for (int nt = 0; nt < 2; ++nt)
#pragma unroll
          for (int g = 0; g < 8; ++g) {
            float pv = __expf(s[mt][nt][g] * scale - mrow[mt][g]);
            rs[g] += pv;
            Ps[wave][mt * 16 + g + hi * 8][nt * 16 + nn] = (bf16_t)pv;
          }
#pragma unroll
        for (int off = 1; off < 16; off <<= 1)
#pragma unroll
          for (int g = 0; g < 8; ++g) rs[g] += __shfl_xor(rs[g], off, 32);
#pragma unroll
        for (int g = 0; g < 8; ++g)
          lrow[mt][g] = lrow[mt][g] * al[g] + rs[g];
#pragma unroll
        for (int dt = 0; dt < 4; ++dt)
#pragma unroll
          for (int g = 0; g < 8; ++g) acc[mt][dt][g] *= al[g];
      }

      // ---- O += P · V (per-wave LDS restage; same-wave DS ops are in order)
      v16bf vf[4];
#pragma unroll
      for (int dt = 0; dt < 4; ++dt)
        vf[dt] = load_frag(&Vt[0][0], 136, dt * 16, sub * 32, lane);
#pragma unroll
      for (int mt = 0; mt < 2; ++mt) {
        v16bf pf = load_frag(&Ps[wave][0][0], 40, mt * 16, 0, lane);
#pragma unroll
        for (int dt = 0; dt < 4; ++dt)
          acc[mt][dt] = wmma_bf16(pf, vf[dt], acc[mt][dt]);
      }
    }
  }

  // ---- epilogue: O = acc / l ----
#pragma unroll
  for (int mt = 0; mt < 2; ++mt)
#pragma unroll
    for (int dt = 0; dt < 4; ++dt)
#pragma unroll
      for (int g = 0; g < 8; ++g) {
        int r = qbase + wave * 32 + mt * 16 + g + hi * 8;
        int c = dcol + dt * 16 + nn;
        O[(size_t)r * 1024 + c] = (bf16_t)(acc[mt][dt][g] / lrow[mt][g]);
      }
}

// ---------------------------------------------------------------------------
extern "C" void kernel_launch(void* const* d_in, const int* in_sizes, int n_in,
                              void* d_out, int out_size, void* d_ws,
                              size_t ws_size, hipStream_t stream) {
  const float* x    = (const float*)d_in[0];  // 4*32*16*16*1024
  const float* Wqkv = (const float*)d_in[1];  // 1024*3072
  const float* Wout = (const float*)d_in[2];  // 1024*1024
  const float* bout = (const float*)d_in[3];  // 1024
  float* out = (float*)d_out;                 // 4*11*16*16*1024 f32

  // Workspace layout (bf16): ~155 MB total.
  bf16_t* WqkvB = (bf16_t*)d_ws;
  bf16_t* WoutB = WqkvB + (size_t)1024 * 3072;
  bf16_t* Qbuf  = WoutB + (size_t)1024 * 1024;
  bf16_t* Kbuf  = Qbuf + (size_t)NQROW * 1024;
  bf16_t* Vbuf  = Kbuf + (size_t)NKVROW * 1024;
  bf16_t* Obuf  = Vbuf + (size_t)NKVROW * 1024;

  cvt_f32_bf16<<<512, 256, 0, stream>>>(Wqkv, WqkvB, 1024 * 3072);
  cvt_f32_bf16<<<256, 256, 0, stream>>>(Wout, WoutB, 1024 * 1024);

  // Q projection: 11264 x 1024
  gemm_bf16<0><<<dim3(NQROW / 128, 1024 / 128), 256, 0, stream>>>(
      x, nullptr, WqkvB, 3072, 0, Qbuf, nullptr, nullptr, nullptr);
  // K/V projection: 24576 x 2048
  gemm_bf16<1><<<dim3(NKVROW / 128, 2048 / 128), 256, 0, stream>>>(
      x, nullptr, WqkvB, 3072, 1024, Kbuf, Vbuf, nullptr, nullptr);
  // Attention: 4*11 slices x 16 heads
  attn_kernel<<<4 * 11 * 16, 256, 0, stream>>>(Qbuf, Kbuf, Vbuf, Obuf);
  // Output projection + bias: 11264 x 1024 (f32 out)
  gemm_bf16<2><<<dim3(NQROW / 128, 1024 / 128), 256, 0, stream>>>(
      nullptr, Obuf, WoutB, 1024, 0, nullptr, nullptr, out, bout);
}
